// PatchSTG_5403068858522
// MI455X (gfx1250) — matrix operand order repaired and checked by reading, not compiled, not run on hardware
//
#include <hip/hip_runtime.h>
#include <hip/hip_bf16.h>
#include <math.h>

// ---------------------------------------------------------------------------
// PatchSTG forward for MI455X (gfx1250).
// - All GEMMs: v_wmma_f32_16x16x32_f16, operands staged in LDS.
// - Weights pre-converted once to f16 N-major panels (prep kernel), then
//   DMA'd into LDS with global_load_async_to_lds_b128 (ASYNCcnt), double-
//   buffered against WMMA phases where LDS budget allows.
// - fp32 residual stream in HBM; LN/softmax via wave32 shuffles.
// ---------------------------------------------------------------------------

typedef __attribute__((ext_vector_type(16))) _Float16 v16h;
typedef __attribute__((ext_vector_type(8)))  float    v8f;

#define B_     16
#define T_     24
#define NODES_ 8192
#define D_     128
#define TPN_   2
#define TP_    12
#define P_     256
#define NPS_   32
#define OUTL_  12
#define LN_EPS 1e-6f
#define PANEL_ 16384  // 128x128 f16 elements

static __device__ __forceinline__ v8f wmma16(v16h a, v16h b, v8f c) {
  return __builtin_amdgcn_wmma_f32_16x16x32_f16(false, a, false, b, (short)0, c,
                                                false, false);
}

// A fragment (16x32 tile of row-major f16, leading dim ld): lane M = lane&15,
// half = lane>>4; elems 0-7 -> K = k0+half*8+j, elems 8-15 -> +16.
static __device__ __forceinline__ v16h load_frag_A(const _Float16* p0, int ld,
                                                   int row0, int k0, int lane) {
  const _Float16* p = p0 + (row0 + (lane & 15)) * ld + k0 + ((lane >> 4) * 8);
  v16h a = {};
#pragma unroll
  for (int j = 0; j < 8; ++j) a[j] = p[j];
#pragma unroll
  for (int j = 0; j < 8; ++j) a[8 + j] = p[16 + j];
  return a;
}

// B fragment from an N-major (N x K row-major) panel: lane column N = lane&15,
// half = lane>>4; elems 0-7 -> K = k0+half*8+j, elems 8-15 -> +16 (matches A).
static __device__ __forceinline__ v16h load_frag_B(const _Float16* p0, int ld,
                                                   int col0, int k0, int lane) {
  const _Float16* p = p0 + (col0 + (lane & 15)) * ld + k0 + ((lane >> 4) * 8);
  v16h b = {};
#pragma unroll
  for (int j = 0; j < 8; ++j) b[j] = p[j];
#pragma unroll
  for (int j = 0; j < 8; ++j) b[8 + j] = p[16 + j];
  return b;
}

// 16x16 tile of A(MxK=128) @ Bt(N-major, ld 128): 4 WMMA K-steps.
static __device__ __forceinline__ v8f tile_mm128(const _Float16* A, int lda,
                                                 const _Float16* Bt, int row0,
                                                 int col0, int lane) {
  v8f acc = {};
#pragma unroll
  for (int k0 = 0; k0 < 128; k0 += 32)
    acc = wmma16(load_frag_A(A, lda, row0, k0, lane),
                 load_frag_B(Bt, 128, col0, k0, lane), acc);
  return acc;
}

static __device__ __forceinline__ float wsum(float v) {
#pragma unroll
  for (int o = 16; o > 0; o >>= 1) v += __shfl_xor(v, o, 32);
  return v;
}
static __device__ __forceinline__ float wmaxf(float v) {
#pragma unroll
  for (int o = 16; o > 0; o >>= 1) v = fmaxf(v, __shfl_xor(v, o, 32));
  return v;
}

// One wave LayerNorms one row of 128 fp32 from global into f16 LDS.
static __device__ __forceinline__ void ln_row(const float* __restrict__ g,
                                              _Float16* l, int lane) {
  float4 v = ((const float4*)g)[lane];
  float s = v.x + v.y + v.z + v.w;
  float q = v.x * v.x + v.y * v.y + v.z * v.z + v.w * v.w;
  s = wsum(s); q = wsum(q);
  float m  = s * (1.0f / 128.0f);
  float iv = rsqrtf(q * (1.0f / 128.0f) - m * m + LN_EPS);
  l[lane * 4 + 0] = (_Float16)((v.x - m) * iv);
  l[lane * 4 + 1] = (_Float16)((v.y - m) * iv);
  l[lane * 4 + 2] = (_Float16)((v.z - m) * iv);
  l[lane * 4 + 3] = (_Float16)((v.w - m) * iv);
}

// Async DMA of one 32KB f16 panel (contiguous in global) into LDS.
// 256 threads x 16B x 8 issues. Caller must wait_async0() + barrier before use.
static __device__ __forceinline__ void async_copy_panel(
    const _Float16* __restrict__ g, _Float16* l, int tid) {
  const unsigned lbase = (unsigned)(size_t)l + (unsigned)tid * 16u;
  const unsigned voff0 = (unsigned)tid * 16u;
  const unsigned long long ga = (unsigned long long)(size_t)g;
#pragma unroll
  for (int i = 0; i < 8; ++i) {
    asm volatile("global_load_async_to_lds_b128 %0, %1, %2"
                 :: "v"(lbase + (unsigned)(i * 4096)),
                    "v"(voff0 + (unsigned)(i * 4096)), "s"(ga)
                 : "memory");
  }
}
static __device__ __forceinline__ void wait_async0() {
  asm volatile("s_wait_asynccnt 0" ::: "memory");
}

// ---------------------------------------------------------------------------
// One-time weight prep: fp32 (K x N row-major) -> f16 N-major 128x128 panels.
// 12 panels per layer: [s_qkv x3][s_proj][s_fc1][s_fc2][n_qkv x3][n_proj]
//                      [n_fc1][n_fc2], 3 layers.
// ---------------------------------------------------------------------------
__global__ void __launch_bounds__(256) patchstg_prep_w(
    const float* __restrict__ sqkv, const float* __restrict__ sprj,
    const float* __restrict__ sf1, const float* __restrict__ sf2,
    const float* __restrict__ nqkv, const float* __restrict__ nprj,
    const float* __restrict__ nf1, const float* __restrict__ nf2,
    _Float16* __restrict__ WP) {
  const int panel = blockIdx.x >> 6;                         // 36 panels
  const int idx = ((blockIdx.x & 63) << 8) + threadIdx.x;    // 0..16383
  const int layer = panel / 12, slot = panel % 12;
  const float* src; int gld, n0;
  switch (slot) {
    case 0: case 1: case 2:
      src = sqkv + (size_t)layer * 128 * 384; gld = 384; n0 = slot * 128; break;
    case 3:  src = sprj + (size_t)layer * PANEL_; gld = 128; n0 = 0; break;
    case 4:  src = sf1  + (size_t)layer * PANEL_; gld = 128; n0 = 0; break;
    case 5:  src = sf2  + (size_t)layer * PANEL_; gld = 128; n0 = 0; break;
    case 6: case 7: case 8:
      src = nqkv + (size_t)layer * 128 * 384; gld = 384; n0 = (slot - 6) * 128; break;
    case 9:  src = nprj + (size_t)layer * PANEL_; gld = 128; n0 = 0; break;
    case 10: src = nf1  + (size_t)layer * PANEL_; gld = 128; n0 = 0; break;
    default: src = nf2  + (size_t)layer * PANEL_; gld = 128; n0 = 0; break;
  }
  const int n = idx >> 7, k = idx & 127;
  WP[(size_t)panel * PANEL_ + idx] = (_Float16)src[k * gld + n0 + n];
}

// ---------------------------------------------------------------------------
// Embedding + patch gather.
// ---------------------------------------------------------------------------
__global__ void __launch_bounds__(256) patchstg_embed(
    const float* __restrict__ x, const int* __restrict__ te,
    const int* __restrict__ reo, const float* __restrict__ w_st,
    const float* __restrict__ b_st, const float* __restrict__ node_emb,
    const float* __restrict__ tod_emb, const float* __restrict__ dow_emb,
    float* __restrict__ X) {
  const int gid = blockIdx.x * 256 + threadIdx.x;
  if (gid >= B_ * TPN_ * NODES_) return;
  const int j  = gid & (NODES_ - 1);
  const int bt = gid >> 13;
  const int t  = bt & (TPN_ - 1);
  const int b  = bt >> 1;
  const int n  = reo[j];

  float out[64];
#pragma unroll
  for (int d = 0; d < 64; ++d) out[d] = b_st[d];
  for (int p = 0; p < TP_; ++p) {
    const int tt = t * TP_ + p;
    const size_t xi = (size_t)(b * T_ + tt) * NODES_ + n;
    const float xv   = x[xi];
    const float todv = (float)te[xi * 2 + 0] * (1.0f / 288.0f);
    const float dowv = (float)te[xi * 2 + 1] * (1.0f / 7.0f);
#pragma unroll
    for (int d = 0; d < 64; ++d) {
      const float* w = w_st + d * 36;  // (64,3,12)
      out[d] += xv * w[p] + todv * w[12 + p] + dowv * w[24 + p];
    }
  }
  float* ob = X + (size_t)gid * D_;
#pragma unroll
  for (int d = 0; d < 64; ++d) ob[d] = out[d];
  const size_t te2 = ((size_t)(b * T_ + (T_ - TPN_) + t) * NODES_ + n) * 2;
  const int tidv = te[te2], diw = te[te2 + 1];
#pragma unroll
  for (int k = 0; k < 16; ++k) ob[64 + k] = tod_emb[tidv * 16 + k];
#pragma unroll
  for (int k = 0; k < 16; ++k) ob[80 + k] = dow_emb[diw * 16 + k];
#pragma unroll
  for (int k = 0; k < 32; ++k) ob[96 + k] = node_emb[n * 32 + k];
}

// ---------------------------------------------------------------------------
// Spatial attention (seq = 32), fused, weight panels double-buffered so the
// async DMA of the next panel overlaps the current WMMA phase.
// Wq panel base = Wqkv; Wk = +PANEL_; Wv = +2*PANEL_; proj panel separate.
// ---------------------------------------------------------------------------
__global__ void __launch_bounds__(256) patchstg_spatial_attn(
    float* __restrict__ X, const _Float16* __restrict__ Wqkv,
    const float* __restrict__ Bqkv, const _Float16* __restrict__ Wp,
    const float* __restrict__ Bp) {
  extern __shared__ char smem[];
  _Float16* sW0  = (_Float16*)smem;            // 128x128 panel (buffer 0)
  _Float16* sW1  = sW0 + PANEL_;               // 128x128 panel (buffer 1)
  _Float16* sXln = sW1 + PANEL_;               // 32x128
  _Float16* sQ   = sXln + 32 * 128;            // 32x128 (pre-scaled)
  _Float16* sK   = sQ + 32 * 128;              // 32x128
  _Float16* sVT  = sK + 32 * 128;              // 128x32 (V transposed)
  _Float16* sCtx = sVT + 128 * 32;             // 32x128
  float*    sSc  = (float*)(sCtx + 32 * 128);  // 32x32 f32
  _Float16* sS   = (_Float16*)(sSc + 32 * 32); // 32x32 f16

  const int tid = threadIdx.x, lane = tid & 31, wave = tid >> 5;
  const int li = lane & 15, half = lane >> 4;
  float* Xb = X + (size_t)blockIdx.x * 32 * D_;
  const float scale = 0.08838834764831845f;  // 1/sqrt(128)

  async_copy_panel(Wqkv, sW0, tid);                       // Wq in flight
#pragma unroll
  for (int i = 0; i < 4; ++i)
    ln_row(Xb + (wave + i * 8) * D_, sXln + (wave + i * 8) * 128, lane);
  wait_async0();
  __syncthreads();

  // ---- Q = LN(x) @ Wq (scaled) ----
  async_copy_panel(Wqkv + PANEL_, sW1, tid);              // Wk in flight
#pragma unroll
  for (int i = 0; i < 2; ++i) {
    const int t = wave + i * 8;
    const int row0 = (t & 1) * 16, col0 = (t >> 1) * 16;
    v8f acc = tile_mm128(sXln, 128, sW0, row0, col0, lane);
    const int n = col0 + li, mh = row0 + 8 * half;
    const float bias = Bqkv[n];
#pragma unroll
    for (int r = 0; r < 8; ++r)
      sQ[(mh + r) * 128 + n] = (_Float16)((acc[r] + bias) * scale);
  }
  wait_async0();
  __syncthreads();

  // ---- K = LN(x) @ Wk ----
  async_copy_panel(Wqkv + 2 * PANEL_, sW0, tid);          // Wv in flight
#pragma unroll
  for (int i = 0; i < 2; ++i) {
    const int t = wave + i * 8;
    const int row0 = (t & 1) * 16, col0 = (t >> 1) * 16;
    v8f acc = tile_mm128(sXln, 128, sW1, row0, col0, lane);
    const int n = col0 + li, mh = row0 + 8 * half;
    const float bias = Bqkv[128 + n];
#pragma unroll
    for (int r = 0; r < 8; ++r) sK[(mh + r) * 128 + n] = (_Float16)(acc[r] + bias);
  }
  wait_async0();
  __syncthreads();

  // ---- V^T = (LN(x) @ Wv)^T ----
  async_copy_panel(Wp, sW1, tid);                         // Wproj in flight
#pragma unroll
  for (int i = 0; i < 2; ++i) {
    const int t = wave + i * 8;
    const int row0 = (t & 1) * 16, col0 = (t >> 1) * 16;
    v8f acc = tile_mm128(sXln, 128, sW0, row0, col0, lane);
    const int n = col0 + li, mh = row0 + 8 * half;
    const float bias = Bqkv[256 + n];
#pragma unroll
    for (int r = 0; r < 8; ++r) sVT[n * 32 + (mh + r)] = (_Float16)(acc[r] + bias);
  }
  __syncthreads();

  // ---- scores = Qs @ K^T ----
  if (wave < 4) {
    const int row0 = (wave & 1) * 16, col0 = (wave >> 1) * 16;
    v8f acc = tile_mm128(sQ, 128, sK, row0, col0, lane);
    const int n = col0 + li, mh = row0 + 8 * half;
#pragma unroll
    for (int r = 0; r < 8; ++r) sSc[(mh + r) * 32 + n] = acc[r];
  }
  __syncthreads();

  // ---- softmax rows ----
#pragma unroll
  for (int i = 0; i < 4; ++i) {
    const int r = wave + i * 8;
    const float v = sSc[r * 32 + lane];
    const float mx = wmaxf(v);
    const float e = __expf(v - mx);
    const float sm = wsum(e);
    sS[r * 32 + lane] = (_Float16)(e / sm);
  }
  __syncthreads();

  // ---- ctx = S @ V (K = 32, single WMMA step) ----
#pragma unroll
  for (int i = 0; i < 2; ++i) {
    const int t = wave + i * 8;
    const int row0 = (t & 1) * 16, col0 = (t >> 1) * 16;
    v8f acc = wmma16(load_frag_A(sS, 32, row0, 0, lane),
                     load_frag_B(sVT, 32, col0, 0, lane), v8f{});
    const int n = col0 + li, mh = row0 + 8 * half;
#pragma unroll
    for (int r = 0; r < 8; ++r) sCtx[(mh + r) * 128 + n] = (_Float16)acc[r];
  }
  wait_async0();
  __syncthreads();

  // ---- out = ctx @ Wp + bp + residual ----
#pragma unroll
  for (int i = 0; i < 2; ++i) {
    const int t = wave + i * 8;
    const int row0 = (t & 1) * 16, col0 = (t >> 1) * 16;
    v8f acc = tile_mm128(sCtx, 128, sW1, row0, col0, lane);
    const int n = col0 + li, mh = row0 + 8 * half;
    const float bias = Bp[n];
#pragma unroll
    for (int r = 0; r < 8; ++r) Xb[(mh + r) * D_ + n] += acc[r] + bias;
  }
}

// ---------------------------------------------------------------------------
// Node attention (seq = 256) over the transposed token view. Whole sequence
// LDS-resident (single weight buffer to fit 320KB/WGP LDS).
// ---------------------------------------------------------------------------
__global__ void __launch_bounds__(256) patchstg_node_attn(
    float* __restrict__ X, const _Float16* __restrict__ Wqkv,
    const float* __restrict__ Bqkv, const _Float16* __restrict__ Wp,
    const float* __restrict__ Bp) {
  extern __shared__ char smem[];
  _Float16* sXln = (_Float16*)smem;      // 256x128; later score scratch
  _Float16* sW   = sXln + 256 * 128;     // 128x128 panel
  _Float16* sQ   = sW + PANEL_;          // 256x128 (rows become ctx)
  _Float16* sK   = sQ + 256 * 128;       // 256x128
  _Float16* sVT  = sK + 256 * 128;       // 128x256 (V transposed)
  float*    sSc  = (float*)smem;         // 32x256 f32 (aliases sXln)
  _Float16* sS   = sXln + 16384;         // 32x256 f16 (aliases sXln)

  const int tid = threadIdx.x, lane = tid & 31, wave = tid >> 5;
  const int li = lane & 15, half = lane >> 4;
  const int bt = blockIdx.x >> 5, s = blockIdx.x & 31;
  const float scale = 0.08838834764831845f;

  async_copy_panel(Wqkv, sW, tid);  // Wq DMA overlaps LayerNorm
#pragma unroll
  for (int i = 0; i < 32; ++i) {
    const int pp = wave + i * 8;
    ln_row(X + ((size_t)(bt * 256 + pp) * 32 + s) * D_, sXln + pp * 128, lane);
  }
  wait_async0();
  __syncthreads();

  // ---- Q (scaled) ----
#pragma unroll
  for (int i = 0; i < 16; ++i) {
    const int t = wave + i * 8;
    const int row0 = (t & 15) * 16, col0 = (t >> 4) * 16;
    v8f acc = tile_mm128(sXln, 128, sW, row0, col0, lane);
    const int n = col0 + li, mh = row0 + 8 * half;
    const float bias = Bqkv[n];
#pragma unroll
    for (int r = 0; r < 8; ++r)
      sQ[(mh + r) * 128 + n] = (_Float16)((acc[r] + bias) * scale);
  }
  __syncthreads();

  // ---- K ----
  async_copy_panel(Wqkv + PANEL_, sW, tid);
  wait_async0();
  __syncthreads();
#pragma unroll
  for (int i = 0; i < 16; ++i) {
    const int t = wave + i * 8;
    const int row0 = (t & 15) * 16, col0 = (t >> 4) * 16;
    v8f acc = tile_mm128(sXln, 128, sW, row0, col0, lane);
    const int n = col0 + li, mh = row0 + 8 * half;
    const float bias = Bqkv[128 + n];
#pragma unroll
    for (int r = 0; r < 8; ++r) sK[(mh + r) * 128 + n] = (_Float16)(acc[r] + bias);
  }
  __syncthreads();

  // ---- V^T ----
  async_copy_panel(Wqkv + 2 * PANEL_, sW, tid);
  wait_async0();
  __syncthreads();
#pragma unroll
  for (int i = 0; i < 16; ++i) {
    const int t = wave + i * 8;
    const int row0 = (t & 15) * 16, col0 = (t >> 4) * 16;
    v8f acc = tile_mm128(sXln, 128, sW, row0, col0, lane);
    const int n = col0 + li, mh = row0 + 8 * half;
    const float bias = Bqkv[256 + n];
#pragma unroll
    for (int r = 0; r < 8; ++r) sVT[n * 256 + (mh + r)] = (_Float16)(acc[r] + bias);
  }
  __syncthreads();

  // ---- attention in 8 chunks of 32 query rows (score scratch aliases sXln) --
  for (int ch = 0; ch < 8; ++ch) {
    const int m0 = ch * 32;
#pragma unroll
    for (int i = 0; i < 4; ++i) {  // scores: 2 x 16 tiles
      const int t = wave + i * 8;
      const int row0 = m0 + (t & 1) * 16, col0 = (t >> 1) * 16;
      v8f acc = tile_mm128(sQ, 128, sK, row0, col0, lane);
      const int n = col0 + li, lm = (t & 1) * 16 + 8 * half;
#pragma unroll
      for (int r = 0; r < 8; ++r) sSc[(lm + r) * 256 + n] = acc[r];
    }
    __syncthreads();
#pragma unroll
    for (int i = 0; i < 4; ++i) {  // softmax rows (256 wide)
      const int r = wave + i * 8;
      float v[8]; float mx = -3.0e38f;
#pragma unroll
      for (int u = 0; u < 8; ++u) {
        v[u] = sSc[r * 256 + lane + 32 * u];
        mx = fmaxf(mx, v[u]);
      }
      mx = wmaxf(mx);
      float sm = 0.f;
#pragma unroll
      for (int u = 0; u < 8; ++u) { v[u] = __expf(v[u] - mx); sm += v[u]; }
      sm = wsum(sm);
      const float inv = 1.0f / sm;
#pragma unroll
      for (int u = 0; u < 8; ++u)
        sS[r * 256 + lane + 32 * u] = (_Float16)(v[u] * inv);
    }
    __syncthreads();
#pragma unroll
    for (int i = 0; i < 2; ++i) {  // ctx chunk -> overwrite consumed Q rows
      const int t = wave + i * 8;
      const int row0 = (t & 1) * 16, col0 = (t >> 1) * 16;
      v8f acc = {};
#pragma unroll
      for (int k0 = 0; k0 < 256; k0 += 32)
        acc = wmma16(load_frag_A(sS, 256, row0, k0, lane),
                     load_frag_B(sVT, 256, col0, k0, lane), acc);
      const int n = col0 + li, mh = m0 + row0 + 8 * half;
#pragma unroll
      for (int r = 0; r < 8; ++r) sQ[(mh + r) * 128 + n] = (_Float16)acc[r];
    }
    __syncthreads();
  }

  // ---- out = ctx @ Wp + bp + residual (strided scatter back) ----
  async_copy_panel(Wp, sW, tid);
  wait_async0();
  __syncthreads();
#pragma unroll
  for (int i = 0; i < 16; ++i) {
    const int t = wave + i * 8;
    const int row0 = (t & 15) * 16, col0 = (t >> 4) * 16;
    v8f acc = tile_mm128(sQ, 128, sW, row0, col0, lane);
    const int n = col0 + li, mh = row0 + 8 * half;
    const float bias = Bp[n];
#pragma unroll
    for (int r = 0; r < 8; ++r) {
      const int pp = mh + r;
      X[((size_t)(bt * 256 + pp) * 32 + s) * D_ + n] += acc[r] + bias;
    }
  }
}

// ---------------------------------------------------------------------------
// Fused MLP: LN -> fc1 -> exact GELU -> fc2 -> residual. Double-buffered
// weight panels (fc2 DMA overlaps fc1 WMMA).  W1 panel = W; W2 = W + PANEL_.
// ---------------------------------------------------------------------------
__global__ void __launch_bounds__(256) patchstg_mlp(
    float* __restrict__ X, const _Float16* __restrict__ W,
    const float* __restrict__ B1, const float* __restrict__ B2) {
  extern __shared__ char smem[];
  _Float16* sW0  = (_Float16*)smem;
  _Float16* sW1  = sW0 + PANEL_;
  _Float16* sXln = sW1 + PANEL_;  // 32x128
  _Float16* sH   = sXln + 32 * 128;

  const int tid = threadIdx.x, lane = tid & 31, wave = tid >> 5;
  const int li = lane & 15, half = lane >> 4;
  float* Xb = X + (size_t)blockIdx.x * 32 * D_;

  async_copy_panel(W, sW0, tid);
#pragma unroll
  for (int i = 0; i < 4; ++i)
    ln_row(Xb + (wave + i * 8) * D_, sXln + (wave + i * 8) * 128, lane);
  wait_async0();
  __syncthreads();

  async_copy_panel(W + PANEL_, sW1, tid);  // fc2 panel in flight
#pragma unroll
  for (int i = 0; i < 2; ++i) {
    const int t = wave + i * 8;
    const int row0 = (t & 1) * 16, col0 = (t >> 1) * 16;
    v8f acc = tile_mm128(sXln, 128, sW0, row0, col0, lane);
    const int n = col0 + li, mh = row0 + 8 * half;
    const float bias = B1[n];
#pragma unroll
    for (int r = 0; r < 8; ++r) {
      const float v = acc[r] + bias;
      const float g = 0.5f * v * (1.0f + erff(v * 0.70710678118654752f));
      sH[(mh + r) * 128 + n] = (_Float16)g;
    }
  }
  wait_async0();
  __syncthreads();

#pragma unroll
  for (int i = 0; i < 2; ++i) {
    const int t = wave + i * 8;
    const int row0 = (t & 1) * 16, col0 = (t >> 1) * 16;
    v8f acc = tile_mm128(sH, 128, sW1, row0, col0, lane);
    const int n = col0 + li, mh = row0 + 8 * half;
    const float bias = B2[n];
#pragma unroll
    for (int r = 0; r < 8; ++r) Xb[(mh + r) * D_ + n] += acc[r] + bias;
  }
}

// ---------------------------------------------------------------------------
// Regression head.
// ---------------------------------------------------------------------------
__global__ void __launch_bounds__(256) patchstg_head(
    const float* __restrict__ X, const int* __restrict__ reo_parts,
    const float* __restrict__ w_reg, const float* __restrict__ b_reg,
    float* __restrict__ out) {
  const int gid = blockIdx.x * 256 + threadIdx.x;
  if (gid >= B_ * OUTL_ * NODES_) return;
  const int n = gid & (NODES_ - 1);
  const int o = (gid >> 13) % OUTL_;
  const int b = gid / (OUTL_ * NODES_);
  const int rp = reo_parts[n];
  float acc = b_reg[o];
#pragma unroll
  for (int t = 0; t < TPN_; ++t) {
    const float* xr = X + ((size_t)(b * TPN_ + t) * NODES_ + rp) * D_;
    const float* wr = w_reg + o * (TPN_ * D_) + t * D_;
    __builtin_prefetch(xr, 0, 0);  // global_prefetch_b8
    for (int d = 0; d < D_; ++d) acc += xr[d] * wr[d];
  }
  out[gid] = acc;
}

// ---------------------------------------------------------------------------
extern "C" void kernel_launch(void* const* d_in, const int* in_sizes, int n_in,
                              void* d_out, int out_size, void* d_ws,
                              size_t ws_size, hipStream_t stream) {
  (void)in_sizes; (void)n_in; (void)out_size; (void)ws_size;

  const float* x         = (const float*)d_in[0];
  const int*   te        = (const int*)d_in[1];
  const int*   reo_all   = (const int*)d_in[2];
  const int*   reo_parts = (const int*)d_in[4];
  const float* w_st      = (const float*)d_in[5];
  const float* b_st      = (const float*)d_in[6];
  const float* node_emb  = (const float*)d_in[7];
  const float* tod_emb   = (const float*)d_in[8];
  const float* dow_emb   = (const float*)d_in[9];
  const float* s_qkv_w = (const float*)d_in[10];
  const float* s_qkv_b = (const float*)d_in[11];
  const float* s_prj_w = (const float*)d_in[12];
  const float* s_prj_b = (const float*)d_in[13];
  const float* s_fc1_w = (const float*)d_in[14];
  const float* s_fc1_b = (const float*)d_in[15];
  const float* s_fc2_w = (const float*)d_in[16];
  const float* s_fc2_b = (const float*)d_in[17];
  const float* n_qkv_w = (const float*)d_in[18];
  const float* n_qkv_b = (const float*)d_in[19];
  const float* n_prj_w = (const float*)d_in[20];
  const float* n_prj_b = (const float*)d_in[21];
  const float* n_fc1_w = (const float*)d_in[22];
  const float* n_fc1_b = (const float*)d_in[23];
  const float* n_fc2_w = (const float*)d_in[24];
  const float* n_fc2_b = (const float*)d_in[25];
  const float* w_reg   = (const float*)d_in[26];
  const float* b_reg   = (const float*)d_in[27];

  const int nTok = B_ * TPN_ * NODES_;           // 262144
  float*     X  = (float*)d_ws;                  // fp32 residual stream (134MB)
  _Float16*  WP = (_Float16*)(X + (size_t)nTok * D_);  // 36 f16 panels (1.2MB)

  patchstg_prep_w<<<36 * 64, 256, 0, stream>>>(s_qkv_w, s_prj_w, s_fc1_w,
                                               s_fc2_w, n_qkv_w, n_prj_w,
                                               n_fc1_w, n_fc2_w, WP);
  patchstg_embed<<<nTok / 256, 256, 0, stream>>>(
      x, te, reo_all, w_st, b_st, node_emb, tod_emb, dow_emb, X);

  const size_t smemS = (size_t)(2 * PANEL_ + 5 * 32 * 128) * 2 +
                       32 * 32 * 4 + 32 * 32 * 2;                // 112,640 B
  const size_t smemM = (size_t)(2 * PANEL_ + 2 * 32 * 128) * 2;  //  81,920 B
  const size_t smemN = (size_t)(3 * 256 * 128 + PANEL_ + 128 * 256) * 2; // 294,912 B

  const int gAttnS = B_ * TPN_ * P_;    // 8192
  const int gMlp   = nTok / 32;         // 8192
  const int gAttnN = B_ * TPN_ * NPS_;  // 1024

  for (int i = 0; i < 3; ++i) {
    const _Float16* lp = WP + (size_t)i * 12 * PANEL_;
    patchstg_spatial_attn<<<gAttnS, 256, smemS, stream>>>(
        X, lp + 0 * PANEL_, s_qkv_b + (size_t)i * 384, lp + 3 * PANEL_,
        s_prj_b + (size_t)i * 128);
    patchstg_mlp<<<gMlp, 256, smemM, stream>>>(
        X, lp + 4 * PANEL_, s_fc1_b + (size_t)i * 128,
        s_fc2_b + (size_t)i * 128);
    patchstg_node_attn<<<gAttnN, 256, smemN, stream>>>(
        X, lp + 6 * PANEL_, n_qkv_b + (size_t)i * 384, lp + 9 * PANEL_,
        n_prj_b + (size_t)i * 128);
    patchstg_mlp<<<gMlp, 256, smemM, stream>>>(
        X, lp + 10 * PANEL_, n_fc1_b + (size_t)i * 128,
        n_fc2_b + (size_t)i * 128);
  }

  patchstg_head<<<(B_ * OUTL_ * NODES_) / 256, 256, 0, stream>>>(
      X, reo_parts, w_reg, b_reg, (float*)d_out);
}
